// Decoder_77068893159870
// MI455X (gfx1250) — matrix operand order repaired and checked by reading, not compiled, not run on hardware
//
#include <hip/hip_runtime.h>

typedef __attribute__((ext_vector_type(16))) _Float16 v16h;
typedef __attribute__((ext_vector_type(8)))  _Float16 v8h;
typedef __attribute__((ext_vector_type(8)))  float    v8f;

#define NBATCH 128
#define LSEQ   1024
#define ROWS_MAIN (NBATCH*LSEQ)      /* 131072 */
#define ROWS_BIG  (NBATCH*LSEQ*3)    /* 393216 */

// ---------------------------------------------------------------------------
// Weight packing: rearrange w[K][64] (K = tap*cin + ci) into the exact
// per-lane B-fragment order used here for v_wmma_f32_16x16x32_f16 (contiguous k, as A), so a B fragment is
// two contiguous 16-byte loads per lane.
//   element e of the v16h B operand = K value (lane>>4)*16 + e, N = nt*16 + (lane&15)
// flat: wp[(((nt*KC)+kc)*32 + lane)*16 + e]
// ---------------------------------------------------------------------------
__global__ void pack_w_kernel(const float* __restrict__ w, _Float16* __restrict__ wp, int KC) {
  int idx = blockIdx.x * 256 + threadIdx.x;
  int total = KC * 2048;
  if (idx >= total) return;
  int e    = idx & 15;
  int lane = (idx >> 4) & 31;
  int q    = idx >> 9;
  int kc   = q % KC;
  int nt   = q / KC;
  int K = kc * 32 + (lane >> 4) * 16 + e;
  int n = nt * 16 + (lane & 15);
  wp[idx] = (_Float16)w[K * 64 + n];
}

__global__ void inv_perm_kernel(const int* __restrict__ perm, int* __restrict__ inv) {
  int idx = blockIdx.x * 256 + threadIdx.x;
  if (idx >= ROWS_MAIN) return;
  int b = idx >> 10;
  int l = idx & 1023;
  inv[(b << 10) + perm[idx]] = l;
}

__global__ void zero_stats_kernel(float* stats) {
  if (threadIdx.x < 128) stats[threadIdx.x] = 0.f;
}

// remap: 1x1 conv, cin=2, cout=64, fused partial BN-statistics
__global__ void remap_conv_kernel(const float* __restrict__ x, const float* __restrict__ w,
                                  const float* __restrict__ bias, _Float16* __restrict__ y,
                                  float* __restrict__ stats) {
  __shared__ float sSum[64], sSsq[64];
  int tid = threadIdx.x;
  if (tid < 64) { sSum[tid] = 0.f; sSsq[tid] = 0.f; }
  __syncthreads();
  size_t idx = (size_t)blockIdx.x * 256 + tid;
  int    n   = (int)(idx & 63);
  size_t m   = idx >> 6;
  float v = x[m * 2] * w[n] + x[m * 2 + 1] * w[64 + n] + bias[n];
  y[idx] = (_Float16)v;
  atomicAdd(&sSum[n], v);
  atomicAdd(&sSsq[n], v * v);
  __syncthreads();
  if (tid < 64) { atomicAdd(&stats[tid], sSum[tid]); atomicAdd(&stats[64 + tid], sSsq[tid]); }
}

__global__ void cvt_f32_f16_kernel(const float* __restrict__ src, _Float16* __restrict__ dst, int n8) {
  int idx = blockIdx.x * 256 + threadIdx.x;
  if (idx >= n8) return;
  const float4* s = (const float4*)(src + (size_t)idx * 8);
  float4 a = s[0], b = s[1];
  v8h o;
  o[0] = (_Float16)a.x; o[1] = (_Float16)a.y; o[2] = (_Float16)a.z; o[3] = (_Float16)a.w;
  o[4] = (_Float16)b.x; o[5] = (_Float16)b.y; o[6] = (_Float16)b.z; o[7] = (_Float16)b.w;
  *(v8h*)(dst + (size_t)idx * 8) = o;
}

// (B, 3L, 64) -> de1/de2/dI, each (B, L, 64); channel block = pos%3
__global__ void split_d_kernel(const _Float16* __restrict__ src, _Float16* __restrict__ de1,
                               _Float16* __restrict__ de2, _Float16* __restrict__ dI) {
  int idx = blockIdx.x * 256 + threadIdx.x;      // ROWS_BIG*8 row-chunks
  if (idx >= ROWS_BIG * 8) return;
  int c8 = idx & 7;
  int r  = idx >> 3;
  int b  = r / (3 * LSEQ);
  int pos = r - b * (3 * LSEQ);
  int l  = pos / 3;
  int which = pos - l * 3;
  _Float16* dst = (which == 0) ? de1 : ((which == 1) ? de2 : dI);
  v8h v = *(const v8h*)(src + (size_t)r * 64 + c8 * 8);
  *(v8h*)(dst + ((size_t)b * LSEQ + l) * 64 + c8 * 8) = v;
}

__global__ void gather_rows_kernel(const _Float16* __restrict__ src, const int* __restrict__ idxmap,
                                   _Float16* __restrict__ dst) {
  int idx = blockIdx.x * 256 + threadIdx.x;      // ROWS_MAIN*8
  if (idx >= ROWS_MAIN * 8) return;
  int c8 = idx & 7;
  int r  = idx >> 3;
  int b  = r >> 10;
  int sl = idxmap[r];
  v8h v = *(const v8h*)(src + (((size_t)b << 10) + sl) * 64 + c8 * 8);
  *(v8h*)(dst + (size_t)r * 64 + c8 * 8) = v;
}

// ---------------------------------------------------------------------------
// WMMA conv core helpers
// ---------------------------------------------------------------------------
__device__ __forceinline__ v16h load_frag16(const _Float16* p) {
  union { v16h v; v8h h[2]; } u;
  u.h[0] = *(const v8h*)(p);
  u.h[1] = *(const v8h*)(p + 8);
  return u.v;
}

__device__ __forceinline__ v8f wmma_f16(const v16h& a, const v16h& b, v8f acc) {
  return __builtin_amdgcn_wmma_f32_16x16x32_f16(false, a, false, b, (short)0, acc, false, false);
}

__device__ __forceinline__ void tile_epilogue(v8f acc, int nt, int lane, size_t rowBase, int mBase,
                                              const float* __restrict__ bias,
                                              _Float16* __restrict__ y,
                                              float* sSum, float* sSsq) {
  int laneM  = lane & 15;
  int laneHi = lane >> 4;
  int n = nt * 16 + laneM;
  float bnn = bias[n];
  float s1 = 0.f, s2 = 0.f;
#pragma unroll
  for (int e = 0; e < 8; ++e) {
    float v = acc[e] + bnn;                 // C/D layout: VGPR e -> M = e + laneHi*8, N = laneM
    s1 += v; s2 += v * v;
    y[(rowBase + (size_t)(mBase + laneHi * 8 + e)) * 64 + n] = (_Float16)v;
  }
  s1 += __shfl_xor(s1, 16, 32);
  s2 += __shfl_xor(s2, 16, 32);
  if (lane < 16) { atomicAdd(&sSum[n], s1); atomicAdd(&sSsq[n], s2); }
}

// ---------------------------------------------------------------------------
// Implicit-GEMM 1D conv (SAME, stride 1) with fused BN statistics.
// CPARTS: number of 64-channel input parts (1 or 3, concatenated logically)
// TAPS:   kernel width (1 or 5).  K = TAPS*64*CPARTS, ordered (tap, channel).
// Optional per-part row gather (perm interleave) folded into LDS staging.
// Block: 128 threads = 4 waves; each wave owns 32 rows x 64 cols
// (two 16x16 M-tiles x four N-tiles), so each B fragment feeds two WMMAs.
// K loop = dynamic tap loop x fully-unrolled chunk loop -> all LDS/global
// fragment accesses use immediate offsets (minimal per-iteration VALU).
// ---------------------------------------------------------------------------
template <int CPARTS, int TAPS>
__global__ void __launch_bounds__(128)
conv_wmma_kernel(const _Float16* __restrict__ in0, const _Float16* __restrict__ in1,
                 const _Float16* __restrict__ in2, const int* __restrict__ gidx, int gflags,
                 const _Float16* __restrict__ wp, const float* __restrict__ bias,
                 _Float16* __restrict__ y, float* __restrict__ stats, int Lseq) {
  constexpr int C     = 64 * CPARTS;
  constexpr int KC    = (TAPS * C) / 32;
  constexpr int CC    = C / 32;               // chunks per tap
  constexpr int TM    = 128;
  constexpr int HALO  = TAPS / 2;
  constexpr int RL    = TM + 2 * HALO;
  constexpr int PITCH = C + 16;               // halves; keeps every fragment 16B aligned
  constexpr int CH8   = C / 8;

  __shared__ __align__(16) _Float16 ldsIn[RL * PITCH];
  __shared__ float sSum[64], sSsq[64];

  const int tid = threadIdx.x;
  if (tid < 64) { sSum[tid] = 0.f; sSsq[tid] = 0.f; }

  const int tilesPerBatch = Lseq / TM;
  const int b  = blockIdx.x / tilesPerBatch;
  const int l0 = (blockIdx.x - b * tilesPerBatch) * TM;
  const size_t rowBase = (size_t)b * Lseq + l0;

  // Stage input tile + halo (zero-padded), optional per-part perm-gather.
  for (int i = tid; i < RL * CH8; i += 128) {
    int r  = i / CH8;
    int c8 = i - r * CH8;
    int l  = l0 + r - HALO;
    v8h val = {};
    if (l >= 0 && l < Lseq) {
      int part = c8 >> 3;
      int pc   = (c8 & 7) * 8;
      int sl = ((gflags >> part) & 1) ? gidx[(size_t)b * Lseq + l] : l;
      const _Float16* src = (part == 0) ? in0 : ((part == 1) ? in1 : in2);
      val = *(const v8h*)(src + ((size_t)b * Lseq + sl) * 64 + pc);
    }
    *(v8h*)(ldsIn + r * PITCH + c8 * 8) = val;
  }
  __syncthreads();

  const int lane   = tid & 31;
  const int wave   = tid >> 5;                // 0..3; 4 waves cover 128 rows
  const int laneM  = lane & 15;
  const int laneHi = lane >> 4;
  const int m0     = wave * 32;

  v8f a00 = {}, a01 = {}, a02 = {}, a03 = {};   // M-tile 0, N-tiles 0..3
  v8f a10 = {}, a11 = {}, a12 = {}, a13 = {};   // M-tile 1, N-tiles 0..3

  constexpr size_t NSTRIDE = (size_t)KC * 32 * 16;   // halves between N-tiles in wp
  const _Float16* arow0 = ldsIn + (m0 + laneM) * PITCH + laneHi * 16;
  const _Float16* arow1 = arow0 + 16 * PITCH;
  const _Float16* bpt   = wp + ((size_t)lane << 4);

  for (int t = 0; t < TAPS; ++t) {
#pragma unroll
    for (int cc = 0; cc < CC; ++cc) {
      // A fragments: two ds_read_b128 per lane per M-tile, immediate offsets
      v16h fa0 = load_frag16(arow0 + cc * 32);
      v16h fa1 = load_frag16(arow1 + cc * 32);
      // B fragments: two contiguous global b128 per lane, immediate offsets;
      // each fragment feeds both M-tiles.
      const _Float16* bp = bpt + (size_t)cc * 512;   // 32 lanes * 16 halves per chunk
      v16h fb0 = load_frag16(bp);
      a00 = wmma_f16(fa0, fb0, a00);  a10 = wmma_f16(fa1, fb0, a10);
      v16h fb1 = load_frag16(bp + NSTRIDE);
      a01 = wmma_f16(fa0, fb1, a01);  a11 = wmma_f16(fa1, fb1, a11);
      v16h fb2 = load_frag16(bp + 2 * NSTRIDE);
      a02 = wmma_f16(fa0, fb2, a02);  a12 = wmma_f16(fa1, fb2, a12);
      v16h fb3 = load_frag16(bp + 3 * NSTRIDE);
      a03 = wmma_f16(fa0, fb3, a03);  a13 = wmma_f16(fa1, fb3, a13);
    }
    arow0 += PITCH;                    // next tap reads input row l+1
    arow1 += PITCH;
    bpt   += (size_t)CC * 512;         // next tap's K-chunks in packed weights
  }

  const int m1 = m0 + 16;
  tile_epilogue(a00, 0, lane, rowBase, m0, bias, y, sSum, sSsq);
  tile_epilogue(a01, 1, lane, rowBase, m0, bias, y, sSum, sSsq);
  tile_epilogue(a02, 2, lane, rowBase, m0, bias, y, sSum, sSsq);
  tile_epilogue(a03, 3, lane, rowBase, m0, bias, y, sSum, sSsq);
  tile_epilogue(a10, 0, lane, rowBase, m1, bias, y, sSum, sSsq);
  tile_epilogue(a11, 1, lane, rowBase, m1, bias, y, sSum, sSsq);
  tile_epilogue(a12, 2, lane, rowBase, m1, bias, y, sSum, sSsq);
  tile_epilogue(a13, 3, lane, rowBase, m1, bias, y, sSum, sSsq);

  __syncthreads();
  if (tid < 64) { atomicAdd(&stats[tid], sSum[tid]); atomicAdd(&stats[64 + tid], sSsq[tid]); }
}

// BN finalize + apply (+ optional ELU)
__global__ void apply_bn_kernel(const _Float16* __restrict__ y, const float* __restrict__ stats,
                                const float* __restrict__ gamma, const float* __restrict__ beta,
                                _Float16* __restrict__ out, int rows, float invCount, int act) {
  __shared__ float sScale[64], sShift[64];
  int tid = threadIdx.x;
  if (tid < 64) {
    float m   = stats[tid] * invCount;
    float var = stats[64 + tid] * invCount - m * m;
    float k   = gamma[tid] * rsqrtf(var + 1.0e-3f);
    sScale[tid] = k;
    sShift[tid] = beta[tid] - m * k;
  }
  __syncthreads();
  int idx = blockIdx.x * 256 + tid;
  if (idx >= rows * 8) return;
  int c0 = (idx & 7) * 8;
  v8h d = *(const v8h*)(y + (size_t)idx * 8);
  v8h o;
#pragma unroll
  for (int j = 0; j < 8; ++j) {
    float t = sScale[c0 + j] * (float)d[j] + sShift[c0 + j];
    if (act) t = (t > 0.f) ? t : (__expf(t) - 1.f);
    o[j] = (_Float16)t;
  }
  *(v8h*)(out + (size_t)idx * 8) = o;
}

// final 1x1 conv 64->2 + softmax
__global__ void out_softmax_kernel(const _Float16* __restrict__ p, const float* __restrict__ w,
                                   const float* __restrict__ bias, float* __restrict__ out) {
  __shared__ float sW[128];
  __shared__ float sB[2];
  int tid = threadIdx.x;
  if (tid < 128) sW[tid] = w[tid];
  if (tid < 2)   sB[tid] = bias[tid];
  __syncthreads();
  int r = blockIdx.x * 256 + tid;
  if (r >= ROWS_MAIN) return;
  float l0 = sB[0], l1 = sB[1];
  const _Float16* row = p + (size_t)r * 64;
#pragma unroll
  for (int c8 = 0; c8 < 8; ++c8) {
    v8h v = *(const v8h*)(row + c8 * 8);
#pragma unroll
    for (int j = 0; j < 8; ++j) {
      float f = (float)v[j];
      l0 += f * sW[(c8 * 8 + j) * 2];
      l1 += f * sW[(c8 * 8 + j) * 2 + 1];
    }
  }
  float mx = fmaxf(l0, l1);
  float e0 = __expf(l0 - mx), e1 = __expf(l1 - mx);
  float s = e0 + e1;
  out[(size_t)r * 2]     = e0 / s;
  out[(size_t)r * 2 + 1] = e1 / s;
}

// ---------------------------------------------------------------------------
extern "C" void kernel_launch(void* const* d_in, const int* in_sizes, int n_in,
                              void* d_out, int out_size, void* d_ws, size_t ws_size,
                              hipStream_t stream) {
  (void)n_in; (void)out_size; (void)ws_size;
  // Input index maps: dict-insertion order vs jax-pytree (alphabetical) order.
  const bool insertionOrder = (in_sizes[0] == NBATCH * 3 * LSEQ * 2);  // flat size of 'x'

  int iX, iPerm, iP0, iOutW, iOutB;
  int wI[12], bI[12], gI[12], eI[12];  // units: 0 remap, 1 demod, 2..6 x1..x5, 7..11 I1..I5
  if (insertionOrder) {
    iX = 0; iPerm = 1; iOutW = 50; iOutB = 51; iP0 = 52;
    auto setU = [&](int u, int s) { wI[u] = s; bI[u] = s + 1; gI[u] = s + 2; eI[u] = s + 3; };
    setU(0, 2); setU(1, 6);
    for (int j = 0; j < 5; ++j) { setU(2 + j, 10 + 8 * j); setU(7 + j, 14 + 8 * j); }
  } else {
    auto setU = [&](int u, int s) { bI[u] = s; eI[u] = s + 1; gI[u] = s + 2; wI[u] = s + 3; };
    for (int j = 0; j < 5; ++j) setU(7 + j, 4 * j);          // I1..I5
    setU(1, 20); iOutB = 24; iOutW = 25; iP0 = 26; setU(0, 27);
    for (int j = 0; j < 5; ++j) setU(2 + j, 31 + 4 * j);     // x1..x5
    iPerm = 51; iX = 52;
  }

  auto F = [&](int i) { return (const float*)d_in[i]; };
  const float* xIn  = F(iX);
  const int*   perm = (const int*)d_in[iPerm];
  const float* p0   = F(iP0);
  const float* outW = F(iOutW);
  const float* outB = F(iOutB);

  // ---- workspace carve (phase-overlapped reuse) ----
  char* base = (char*)d_ws;
  size_t off = 0;
  auto carve = [&](size_t bytes) -> char* {
    char* p = base + off;
    off += (bytes + 255) & ~(size_t)255;
    return p;
  };
  float* stats = (float*)carve(128 * sizeof(float));
  int*   inv   = (int*)carve((size_t)ROWS_MAIN * sizeof(int));
  _Float16* wpDemod = (_Float16*)carve((size_t)2 * 2048 * sizeof(_Float16));
  _Float16* wpX[5]; _Float16* wpI[5];
  for (int j = 0; j < 5; ++j) wpX[j] = (_Float16*)carve((size_t)((j == 0) ? 30 : 10) * 2048 * sizeof(_Float16));
  for (int j = 0; j < 5; ++j) wpI[j] = (_Float16*)carve((size_t)((j == 0) ? 30 : 10) * 2048 * sizeof(_Float16));
  const size_t BIG = (size_t)ROWS_BIG * 64 * sizeof(_Float16);  // 50.3 MB
  char* regA = carve(BIG);  // d_remap, then de1/de2/dI
  char* regB = carve(BIG);  // d_demod, then p / bufA / bufB
  char* regY = carve(BIG);  // pre-BN conv scratch
  _Float16* dRemap = (_Float16*)regA;
  _Float16* dDemod = (_Float16*)regB;
  _Float16* yScr   = (_Float16*)regY;
  const size_t THIRD = (size_t)ROWS_MAIN * 64;
  _Float16* de1  = (_Float16*)regA;
  _Float16* de2  = de1 + THIRD;
  _Float16* dIb  = de2 + THIRD;
  _Float16* pBuf = (_Float16*)regB;
  _Float16* bufA = pBuf + THIRD;
  _Float16* bufB = bufA + THIRD;

  // ---- pack weights into B-fragment order (deterministic, every call) ----
  auto packW = [&](int unit, _Float16* wp, int KC) {
    int total = KC * 2048;
    pack_w_kernel<<<(total + 255) / 256, 256, 0, stream>>>(F(wI[unit]), wp, KC);
  };
  packW(1, wpDemod, 2);
  for (int j = 0; j < 5; ++j) packW(2 + j, wpX[j], (j == 0) ? 30 : 10);
  for (int j = 0; j < 5; ++j) packW(7 + j, wpI[j], (j == 0) ? 30 : 10);

  inv_perm_kernel<<<ROWS_MAIN / 256, 256, 0, stream>>>(perm, inv);

  // ---- remap unit (1x1, cin=2) ----
  zero_stats_kernel<<<1, 128, 0, stream>>>(stats);
  remap_conv_kernel<<<(ROWS_BIG * 64) / 256, 256, 0, stream>>>(xIn, F(wI[0]), F(bI[0]), yScr, stats);
  apply_bn_kernel<<<ROWS_BIG / 32, 256, 0, stream>>>(yScr, stats, F(gI[0]), F(eI[0]), dRemap,
                                                     ROWS_BIG, 1.f / (float)ROWS_BIG, 1);
  // ---- demod unit (1x1, WMMA, linear act) ----
  zero_stats_kernel<<<1, 128, 0, stream>>>(stats);
  conv_wmma_kernel<1, 1><<<ROWS_BIG / 128, 128, 0, stream>>>(dRemap, dRemap, dRemap, nullptr, 0,
                                                             wpDemod, F(bI[1]), yScr, stats, 3 * LSEQ);
  apply_bn_kernel<<<ROWS_BIG / 32, 256, 0, stream>>>(yScr, stats, F(gI[1]), F(eI[1]), dDemod,
                                                     ROWS_BIG, 1.f / (float)ROWS_BIG, 0);
  // ---- reshape split + p0 cast ----
  split_d_kernel<<<(ROWS_BIG * 8) / 256, 256, 0, stream>>>(dDemod, de1, de2, dIb);
  cvt_f32_f16_kernel<<<(ROWS_MAIN * 64 / 8) / 256, 256, 0, stream>>>(p0, pBuf, ROWS_MAIN * 64 / 8);

  const float invCntMain = 1.f / (float)ROWS_MAIN;
  auto unit192 = [&](const _Float16* a0, const _Float16* a1, const _Float16* a2,
                     const int* gi, int gf, _Float16* wp, int unit, _Float16* dst) {
    zero_stats_kernel<<<1, 128, 0, stream>>>(stats);
    conv_wmma_kernel<3, 5><<<ROWS_MAIN / 128, 128, 0, stream>>>(a0, a1, a2, gi, gf, wp,
                                                                F(bI[unit]), yScr, stats, LSEQ);
    apply_bn_kernel<<<ROWS_MAIN / 32, 256, 0, stream>>>(yScr, stats, F(gI[unit]), F(eI[unit]),
                                                        dst, ROWS_MAIN, invCntMain, 1);
  };
  auto unit64 = [&](const _Float16* a0, _Float16* wp, int unit, _Float16* dst) {
    zero_stats_kernel<<<1, 128, 0, stream>>>(stats);
    conv_wmma_kernel<1, 5><<<ROWS_MAIN / 128, 128, 0, stream>>>(a0, a0, a0, nullptr, 0, wp,
                                                                F(bI[unit]), yScr, stats, LSEQ);
    apply_bn_kernel<<<ROWS_MAIN / 32, 256, 0, stream>>>(yScr, stats, F(gI[unit]), F(eI[unit]),
                                                        dst, ROWS_MAIN, invCntMain, 1);
  };

  for (int it = 0; it < 5; ++it) {
    // x-chain: q = [de2, p, de1]
    unit192(de2, pBuf, de1, nullptr, 0, wpX[0], 2, bufA);
    _Float16* cur = bufA; _Float16* oth = bufB;
    for (int j = 1; j < 5; ++j) { unit64(cur, wpX[j], 2 + j, oth); _Float16* t = cur; cur = oth; oth = t; }
    // cur == bufA (x5 output)
    // I-chain: q = [dI, interleave(q), interleave(de1)]  (parts 1,2 gathered by perm)
    unit192(dIb, cur, de1, perm, 6, wpI[0], 7, oth);
    cur = oth; oth = (cur == bufA) ? bufB : bufA;
    for (int j = 1; j < 5; ++j) { unit64(cur, wpI[j], 7 + j, oth); _Float16* t = cur; cur = oth; oth = t; }
    // p = deinterleave(q) via inv
    gather_rows_kernel<<<(ROWS_MAIN * 8) / 256, 256, 0, stream>>>(cur, inv, pBuf);
  }

  out_softmax_kernel<<<(ROWS_MAIN + 255) / 256, 256, 0, stream>>>(pBuf, outW, outB, (float*)d_out);
}
